// EncoderLayer_46076409152424
// MI455X (gfx1250) — matrix operand order repaired and checked by reading, not compiled, not run on hardware
//
#include <hip/hip_runtime.h>
#include <math.h>

// ---------------------------------------------------------------------------
// MI455X (gfx1250) encoder layer. All GEMMs via V_WMMA_F32_16X16X32_F16,
// f32 operands converted to f16 (v_cvt_pk_rtz_f16_f32) while staging through
// LDS, f32 accumulate. Tiles: 128x64x32, 128 threads (4 waves), each wave =
// 32x64 strip (2x4 WMMA), double-buffered LDS with a peeled software pipeline,
// float4 global loads, b64 LDS stores.
// Host guarantees: M % 128 == 0, K % 32 == 0, N % 4 == 0, all pointers/leading
// dims 16B-aligned. Only N may be partial vs. the 64-wide tile (attn@V, N=32).
// ---------------------------------------------------------------------------

typedef __attribute__((ext_vector_type(16))) _Float16 v16h;
typedef __attribute__((ext_vector_type(2)))  __fp16   v2fp16;  // cvt_pkrtz return type
typedef __attribute__((ext_vector_type(8)))  float    v8f;

static constexpr int TBM = 128, TBN = 64, TBK = 32;
static constexpr int LDSH = 36;   // f16 row stride: 72B (8B aligned), 18 dwords, conflict-free

struct GemmP {
  const float* A;  long lda, sAo, sAi;      // A offset = (z/zdiv)*sAo + (z%zdiv)*sAi
  const float* Bp; long ldb, sBo, sBi;
  float*       C;  long ldc, sCo, sCi;
  int M, N, K, zdiv;
  const float* bias;                         // per-N bias (nullable)
  const float* cadd;   int cadd_mod;         // scalar add, index z%cadd_mod (nullable)
  const float* ascale; int ascale_mod; long ascale_stride; // per-K A scale (nullable)
  int b_nk;                                  // 1: B memory is [N][K] (i.e. A@B^T)
  int act;                                   // 0 none, 1 relu, 2 gelu(exact)
};

union FragU { v16h h; unsigned u[8]; };
union H4U  { v2fp16 h2[2]; uint2 v; };

// float4 -> 4xf16 packed in one uint2 via two v_cvt_pk_rtz_f16_f32
__device__ __forceinline__ uint2 cvt4(float4 a) {
  H4U t;
  t.h2[0] = __builtin_amdgcn_cvt_pkrtz(a.x, a.y);
  t.h2[1] = __builtin_amdgcn_cvt_pkrtz(a.z, a.w);
  return t.v;
}

// A fragment: 16x32 f16. lane L: M = base + (L&15).
// dword v: K-pair = (v<4 ? (L>>4)*8 + 2v : 16 + (L>>4)*8 + 2(v-4))   [ISA 7.12.2]
__device__ __forceinline__ v16h load_afrag(const _Float16* As, int mbase, int lane) {
  FragU f;
  const int half = lane >> 4, r = lane & 15;
  const unsigned* row = (const unsigned*)As + (mbase + r) * (LDSH / 2);
#pragma unroll
  for (int v = 0; v < 8; ++v) {
    const int d = ((v >= 4) ? 8 : 0) + half * 4 + (v & 3);
    f.u[v] = row[d];
  }
  return f.h;
}

// B fragment: 32x16 f16 stored transposed in LDS as Bs[n][k].
// lane L: N = base + (L&15); dword v: K-pair = (v<4 ? (L>>4)*8 + 2v : 16 + (L>>4)*8 + 2(v-4)).
__device__ __forceinline__ v16h load_bfrag(const _Float16* Bs, int nbase, int lane) {
  FragU f;
  const int half = lane >> 4, r = lane & 15;
  const unsigned* row = (const unsigned*)Bs + (nbase + r) * (LDSH / 2);
#pragma unroll
  for (int v = 0; v < 8; ++v) f.u[v] = row[((v >= 4) ? 8 : 0) + half * 4 + (v & 3)];
  return f.h;
}

template <bool BNK, bool ASC>
__global__ __launch_bounds__(128)
void gemm_wmma_f16(GemmP p) {
  __shared__ __attribute__((aligned(16))) _Float16 As[2][TBM * LDSH];
  __shared__ __attribute__((aligned(16))) _Float16 Bs[2][TBN * LDSH];

  const int tid = threadIdx.x;
  const int z  = blockIdx.z;
  const int zq = z / p.zdiv, zr = z % p.zdiv;
  const float* A  = p.A  + (long)zq * p.sAo + (long)zr * p.sAi;
  const float* Bg = p.Bp + (long)zq * p.sBo + (long)zr * p.sBi;
  float*       C  = p.C  + (long)zq * p.sCo + (long)zr * p.sCi;
  const float* asc = ASC ? (p.ascale + (long)(z % p.ascale_mod) * p.ascale_stride) : nullptr;
  const float caddv = p.cadd ? p.cadd[z % p.cadd_mod] : 0.f;

  const int m0 = blockIdx.y * TBM;
  const int n0 = blockIdx.x * TBN;

  const int lane = tid & 31;
  const int wave = tid >> 5;
  const int wm = wave * 32;               // each wave: rows [wm, wm+32) x all 64 cols

  // register staging: A 128x32 -> 8 float4/thread; B 64x32 -> 4 float4/thread
  const int aRow = tid >> 3;              // 0..15
  const int aCol = (tid & 7) * 4;         // 0,4,..,28
  const int bK   = tid >> 4;              // 0..7   (transpose path: k row)
  const int bN   = (tid & 15) * 4;        // 0..60  (transpose path: n col)
  const bool bOk = (n0 + bN) < p.N;       // whole float4 in/out (N % 4 == 0)

  float4 aR[8], bR[4];

  auto loadTiles = [&](int k0) {
#pragma unroll
    for (int e = 0; e < 8; ++e) {
      float4 v = *(const float4*)(A + (long)(m0 + e * 16 + aRow) * p.lda + (k0 + aCol));
      if (ASC) {
        const float* s = asc + k0 + aCol;
        v.x *= s[0]; v.y *= s[1]; v.z *= s[2]; v.w *= s[3];
      }
      aR[e] = v;
    }
    if (BNK) {
#pragma unroll
      for (int e = 0; e < 4; ++e)
        bR[e] = *(const float4*)(Bg + (long)(n0 + e * 16 + aRow) * p.ldb + (k0 + aCol));
    } else {
#pragma unroll
      for (int e = 0; e < 4; ++e) {
        float4 v = make_float4(0.f, 0.f, 0.f, 0.f);
        if (bOk) v = *(const float4*)(Bg + (long)(k0 + e * 8 + bK) * p.ldb + (n0 + bN));
        bR[e] = v;
      }
    }
  };

  auto stageTiles = [&](int buf) {
#pragma unroll
    for (int e = 0; e < 8; ++e)
      *(uint2*)&As[buf][(e * 16 + aRow) * LDSH + aCol] = cvt4(aR[e]);
    if (BNK) {
#pragma unroll
      for (int e = 0; e < 4; ++e)
        *(uint2*)&Bs[buf][(e * 16 + aRow) * LDSH + aCol] = cvt4(bR[e]);
    } else {
#pragma unroll
      for (int e = 0; e < 4; ++e) {
        const int k = e * 8 + bK;
        Bs[buf][(bN + 0) * LDSH + k] = (_Float16)bR[e].x;
        Bs[buf][(bN + 1) * LDSH + k] = (_Float16)bR[e].y;
        Bs[buf][(bN + 2) * LDSH + k] = (_Float16)bR[e].z;
        Bs[buf][(bN + 3) * LDSH + k] = (_Float16)bR[e].w;
      }
    }
  };

  v8f acc[2][4] = {};

  auto computeStep = [&](int buf) {
    v16h af[2], bf[4];
    af[0] = load_afrag(&As[buf][0], wm,      lane);
    af[1] = load_afrag(&As[buf][0], wm + 16, lane);
#pragma unroll
    for (int nb = 0; nb < 4; ++nb) bf[nb] = load_bfrag(&Bs[buf][0], nb * 16, lane);
#pragma unroll
    for (int mb = 0; mb < 2; ++mb)
#pragma unroll
      for (int nb = 0; nb < 4; ++nb)
        acc[mb][nb] = __builtin_amdgcn_wmma_f32_16x16x32_f16(
            false, af[mb], false, bf[nb], (short)0, acc[mb][nb], false, false);
  };

  const int nt = p.K / TBK;
  loadTiles(0);
  stageTiles(0);
  __syncthreads();
  int cur = 0;

  // peeled pipeline: branch-free steady state, last compute outside the loop
  for (int t = 0; t + 1 < nt; ++t) {
    loadTiles((t + 1) * TBK);     // next tile's global_load_b128s in flight
    computeStep(cur);             // 8x WMMA on current LDS buffer
    stageTiles(cur ^ 1);
    __syncthreads();
    cur ^= 1;
  }
  computeStep(cur);

  // epilogue: C/D layout per ISA: VGPR r -> M = (lane>=16 ? 8 : 0) + r, N = lane&15
  const int half = lane >> 4, lr = lane & 15;
#pragma unroll
  for (int nb = 0; nb < 4; ++nb) {
    const int gcol = n0 + nb * 16 + lr;
    if (gcol >= p.N) continue;
    const float bv = p.bias ? p.bias[gcol] : 0.f;
#pragma unroll
    for (int mb = 0; mb < 2; ++mb) {
#pragma unroll
      for (int r = 0; r < 8; ++r) {
        const int grow = m0 + wm + mb * 16 + half * 8 + r;   // M always full tile
        float v = acc[mb][nb][r] + bv + caddv;
        if (p.act == 1)      v = fmaxf(v, 0.f);
        else if (p.act == 2) v = 0.5f * v * (1.f + erff(v * 0.70710678118654752f));
        C[(long)grow * p.ldc + gcol] = v;
      }
    }
  }
}

// ---------------------------------------------------------------------------
// Elementwise / reduction kernels
// ---------------------------------------------------------------------------

static constexpr float kInvSqrtDep = 0.17677669529663687f;  // 1/sqrt(32)

// scores(B,H,S,S) : local branch = s/sqrt(d) + mask*-1e9 + adjoin
__global__ void score_fix_local(float* sc, const int* mask, const float* adjoin) {
  const unsigned idx = blockIdx.x * 256u + threadIdx.x;
  const int j = idx & 511, i = (idx >> 9) & 511, b = idx >> 21;
  float s = sc[idx] * kInvSqrtDep;
  s += (float)mask[b * 512 + j] * -1e9f;
  s += adjoin[(size_t)b * 262144 + i * 512 + j];
  sc[idx] = s;
}

// global branch = relu(s)*((1+e)/(1+exp(1-dist)))/sqrt(d) + mask*-1e9
__global__ void score_fix_global(float* sc, const int* mask, const float* dist) {
  const unsigned idx = blockIdx.x * 256u + threadIdx.x;
  const int j = idx & 511, i = (idx >> 9) & 511, b = idx >> 21;
  const float w = dist[(size_t)b * 262144 + i * 512 + j];
  const float resc = 3.7182818284590452f / (1.f + expf(1.f - w));
  float s = fmaxf(sc[idx], 0.f) * resc * kInvSqrtDep;
  s += (float)mask[b * 512 + j] * -1e9f;
  sc[idx] = s;
}

__global__ __launch_bounds__(128) void softmax512(float* sc) {
  __shared__ float red[128];
  float* p = sc + (size_t)blockIdx.x * 512;
  const int t = threadIdx.x;
  float v0 = p[t], v1 = p[t + 128], v2 = p[t + 256], v3 = p[t + 384];
  float m = fmaxf(fmaxf(v0, v1), fmaxf(v2, v3));
  red[t] = m; __syncthreads();
  for (int s = 64; s > 0; s >>= 1) { if (t < s) red[t] = fmaxf(red[t], red[t + s]); __syncthreads(); }
  m = red[0]; __syncthreads();
  v0 = expf(v0 - m); v1 = expf(v1 - m); v2 = expf(v2 - m); v3 = expf(v3 - m);
  red[t] = v0 + v1 + v2 + v3; __syncthreads();
  for (int s = 64; s > 0; s >>= 1) { if (t < s) red[t] += red[t + s]; __syncthreads(); }
  const float inv = 1.f / red[0];
  p[t] = v0 * inv; p[t + 128] = v1 * inv; p[t + 256] = v2 * inv; p[t + 384] = v3 * inv;
}

// logits = v_*(hq0+hq1); batchnorm; *concat(x_l,x_g); +x; LayerNorm(ln1)
__global__ __launch_bounds__(256)
void ban_ln1(const float* v_, const float* hq0, const float* hq1,
             const float* xl, const float* xg, const float* x,
             const float* bg, const float* bb, const float* bm, const float* bv,
             const float* g1, const float* b1, float* out1) {
  __shared__ float red[256];
  const size_t r = blockIdx.x;
  const int t = threadIdx.x;
  float y[2];
#pragma unroll
  for (int e = 0; e < 2; ++e) {
    const int c = t + e * 256;
    const size_t idx = r * 512 + c;
    float lg = v_[idx] * (hq0[idx] + hq1[idx]);
    lg = (lg - bm[c]) * rsqrtf(bv[c] + 1e-3f) * bg[c] + bb[c];
    const float cc = (c < 256) ? xl[r * 256 + c] : xg[r * 256 + (c - 256)];
    y[e] = x[idx] + lg * cc;
  }
  red[t] = y[0] + y[1]; __syncthreads();
  for (int s = 128; s > 0; s >>= 1) { if (t < s) red[t] += red[t + s]; __syncthreads(); }
  const float mu = red[0] * (1.f / 512.f); __syncthreads();
  const float d0 = y[0] - mu, d1 = y[1] - mu;
  red[t] = d0 * d0 + d1 * d1; __syncthreads();
  for (int s = 128; s > 0; s >>= 1) { if (t < s) red[t] += red[t + s]; __syncthreads(); }
  const float inv = rsqrtf(red[0] * (1.f / 512.f) + 1e-6f);
  out1[r * 512 + t]       = d0 * inv * g1[t]       + b1[t];
  out1[r * 512 + t + 256] = d1 * inv * g1[t + 256] + b1[t + 256];
}

// out = LayerNorm(a + b)
__global__ __launch_bounds__(256)
void add_ln(const float* a, const float* bsrc, const float* g, const float* bb, float* outp) {
  __shared__ float red[256];
  const size_t r = blockIdx.x;
  const int t = threadIdx.x;
  const float y0 = a[r * 512 + t] + bsrc[r * 512 + t];
  const float y1 = a[r * 512 + t + 256] + bsrc[r * 512 + t + 256];
  red[t] = y0 + y1; __syncthreads();
  for (int s = 128; s > 0; s >>= 1) { if (t < s) red[t] += red[t + s]; __syncthreads(); }
  const float mu = red[0] * (1.f / 512.f); __syncthreads();
  const float d0 = y0 - mu, d1 = y1 - mu;
  red[t] = d0 * d0 + d1 * d1; __syncthreads();
  for (int s = 128; s > 0; s >>= 1) { if (t < s) red[t] += red[t + s]; __syncthreads(); }
  const float inv = rsqrtf(red[0] * (1.f / 512.f) + 1e-6f);
  outp[r * 512 + t]       = d0 * inv * g[t]       + bb[t];
  outp[r * 512 + t + 256] = d1 * inv * g[t + 256] + bb[t + 256];
}

// ---------------------------------------------------------------------------
// Host orchestration
// ---------------------------------------------------------------------------

static void launch_gemm(const GemmP& p, int batches, hipStream_t stream) {
  dim3 g((p.N + TBN - 1) / TBN, (p.M + TBM - 1) / TBM, batches);
  if (p.b_nk) {
    if (p.ascale) gemm_wmma_f16<true,  true ><<<g, dim3(128), 0, stream>>>(p);
    else          gemm_wmma_f16<true,  false><<<g, dim3(128), 0, stream>>>(p);
  } else {
    if (p.ascale) gemm_wmma_f16<false, true ><<<g, dim3(128), 0, stream>>>(p);
    else          gemm_wmma_f16<false, false><<<g, dim3(128), 0, stream>>>(p);
  }
}

extern "C" void kernel_launch(void* const* d_in, const int* in_sizes, int n_in,
                              void* d_out, int out_size, void* d_ws, size_t ws_size,
                              hipStream_t stream) {
  (void)in_sizes; (void)n_in; (void)out_size; (void)ws_size;

  const float* x      = (const float*)d_in[0];
  const float* adjoin = (const float*)d_in[1];
  const float* dist   = (const float*)d_in[2];
  const float* wq1 = (const float*)d_in[3],  *bq1 = (const float*)d_in[4];
  const float* wk1 = (const float*)d_in[5],  *bk1 = (const float*)d_in[6];
  const float* wv1 = (const float*)d_in[7],  *bv1 = (const float*)d_in[8];
  const float* wo1 = (const float*)d_in[9],  *bo1 = (const float*)d_in[10];
  const float* wq2 = (const float*)d_in[11], *bq2 = (const float*)d_in[12];
  const float* wk2 = (const float*)d_in[13], *bk2 = (const float*)d_in[14];
  const float* wv2 = (const float*)d_in[15], *bv2 = (const float*)d_in[16];
  const float* wo2 = (const float*)d_in[17], *bo2 = (const float*)d_in[18];
  const float* vnet_w = (const float*)d_in[19], *vnet_b = (const float*)d_in[20];
  const float* qnet_w = (const float*)d_in[21], *qnet_b = (const float*)d_in[22];
  const float* h_mat  = (const float*)d_in[23], *h_bias = (const float*)d_in[24];
  const float* bn_g = (const float*)d_in[25], *bn_b = (const float*)d_in[26];
  const float* bn_m = (const float*)d_in[27], *bn_v = (const float*)d_in[28];
  const float* ln1g = (const float*)d_in[29], *ln1b = (const float*)d_in[30];
  const float* ln2g = (const float*)d_in[31], *ln2b = (const float*)d_in[32];
  const float* fw1 = (const float*)d_in[33], *fb1 = (const float*)d_in[34];
  const float* fw2 = (const float*)d_in[35], *fb2 = (const float*)d_in[36];
  const int*   maskI = (const int*)d_in[37];

  float* out  = (float*)d_out;
  float* aw_l = out + 2097152;        // (B,H,S,S)
  float* aw_g = out + 18874368;

  float* ws = (float*)d_ws;
  float* q1  = ws;                float* k1  = ws + 1048576;  float* v1 = ws + 2097152;
  float* q2  = ws + 3145728;      float* k2  = ws + 4194304;  float* v2 = ws + 5242880;
  float* av1 = ws + 6291456;      float* av2 = ws + 7340032;
  float* xl  = ws + 8388608;      float* xg  = ws + 9437184;
  float* vv  = ws + 10485760;     float* qq  = ws + 12582912;   // v_, q_ : (B,S,512)
  float* att = ws + 14680064;                                   // (B,2,S,S)
  float* hq0 = ws + 18874368;     float* hq1 = ws + 20971520;
  float* out1 = ws + 23068672;
  float* ffnh = ws + 14680064;    // reuses att/hq0/hq1 (dead by FFN time), 8388608 floats
  float* ffno = ws + 25165824;

  // ---- 1) QKV projections (both branches): (4096,256) @ (256,256) + bias
  struct ProjCfg { const float* a; const float* w; const float* b; float* o; };
  const ProjCfg proj[6] = {
    {x,       wq1, bq1, q1}, {x,       wk1, bk1, k1}, {x,       wv1, bv1, v1},
    {x + 256, wq2, bq2, q2}, {x + 256, wk2, bk2, k2}, {x + 256, wv2, bv2, v2}};
  for (int i = 0; i < 6; ++i) {
    GemmP p{}; p.A = proj[i].a; p.lda = 512; p.Bp = proj[i].w; p.ldb = 256;
    p.C = proj[i].o; p.ldc = 256; p.M = 4096; p.N = 256; p.K = 256; p.zdiv = 1;
    p.bias = proj[i].b;
    launch_gemm(p, 1, stream);
  }

  // ---- 2) scores = q @ k^T, batched over (b,h) -> directly into d_out aw regions
  for (int br = 0; br < 2; ++br) {
    GemmP p{};
    p.A = br ? q2 : q1; p.lda = 256; p.sAo = 131072; p.sAi = 32;
    p.Bp = br ? k2 : k1; p.ldb = 256; p.sBo = 131072; p.sBi = 32; p.b_nk = 1;
    p.C = br ? aw_g : aw_l; p.ldc = 512; p.sCo = 2097152; p.sCi = 262144;
    p.M = 512; p.N = 512; p.K = 32; p.zdiv = 8;
    launch_gemm(p, 64, stream);
  }

  // ---- 3) mask / bias / dist-rescale, then softmax (in place in d_out)
  score_fix_local <<<65536, 256, 0, stream>>>(aw_l, maskI, adjoin);
  score_fix_global<<<65536, 256, 0, stream>>>(aw_g, maskI, dist);
  softmax512<<<32768, 128, 0, stream>>>(aw_l);
  softmax512<<<32768, 128, 0, stream>>>(aw_g);

  // ---- 4) attn @ V -> (B,S,DH) laid out head-interleaved
  for (int br = 0; br < 2; ++br) {
    GemmP p{};
    p.A = br ? aw_g : aw_l; p.lda = 512; p.sAo = 2097152; p.sAi = 262144;
    p.Bp = br ? v2 : v1; p.ldb = 256; p.sBo = 131072; p.sBi = 32;
    p.C = br ? av2 : av1; p.ldc = 256; p.sCo = 131072; p.sCi = 32;
    p.M = 512; p.N = 32; p.K = 512; p.zdiv = 8;
    launch_gemm(p, 64, stream);
  }

  // ---- 5) output projections: x_l / x_g
  {
    GemmP p{}; p.A = av1; p.lda = 256; p.Bp = wo1; p.ldb = 256; p.C = xl; p.ldc = 256;
    p.M = 4096; p.N = 256; p.K = 256; p.zdiv = 1; p.bias = bo1; launch_gemm(p, 1, stream);
    GemmP q{}; q.A = av2; q.lda = 256; q.Bp = wo2; q.ldb = 256; q.C = xg; q.ldc = 256;
    q.M = 4096; q.N = 256; q.K = 256; q.zdiv = 1; q.bias = bo2; launch_gemm(q, 1, stream);
  }

  // ---- 6) v_ = relu(x_l@vnet_w+b), q_ = relu(x_g@qnet_w+b) : (4096,256)@(256,512)
  {
    GemmP p{}; p.A = xl; p.lda = 256; p.Bp = vnet_w; p.ldb = 512; p.C = vv; p.ldc = 512;
    p.M = 4096; p.N = 512; p.K = 256; p.zdiv = 1; p.bias = vnet_b; p.act = 1; launch_gemm(p, 1, stream);
    GemmP q{}; q.A = xg; q.lda = 256; q.Bp = qnet_w; q.ldb = 512; q.C = qq; q.ldc = 512;
    q.M = 4096; q.N = 512; q.K = 256; q.zdiv = 1; q.bias = qnet_b; q.act = 1; launch_gemm(q, 1, stream);
  }

  // ---- 7) att_maps[b,h] = (v_ * h_mat[h]) @ q_^T + h_bias[h]   (batch z = b*2+h)
  {
    GemmP p{};
    p.A = vv; p.lda = 512; p.sAo = 262144; p.sAi = 0;
    p.ascale = h_mat; p.ascale_mod = 2; p.ascale_stride = 512;
    p.Bp = qq; p.ldb = 512; p.sBo = 262144; p.sBi = 0; p.b_nk = 1;
    p.C = att; p.ldc = 512; p.sCo = 524288; p.sCi = 262144;
    p.cadd = h_bias; p.cadd_mod = 2;
    p.M = 512; p.N = 512; p.K = 512; p.zdiv = 2;
    launch_gemm(p, 16, stream);
  }

  // ---- 8) hq_h[b] = att_maps[b,h] @ q_[b]   (sum over h done in ban_ln1)
  for (int h = 0; h < 2; ++h) {
    GemmP p{};
    p.A = att + (long)h * 262144; p.lda = 512; p.sAo = 524288;
    p.Bp = qq; p.ldb = 512; p.sBo = 262144;
    p.C = h ? hq1 : hq0; p.ldc = 512; p.sCo = 262144;
    p.M = 512; p.N = 512; p.K = 512; p.zdiv = 1;
    launch_gemm(p, 8, stream);
  }

  // ---- 9) BAN logits -> batchnorm -> gate -> residual -> LayerNorm1
  ban_ln1<<<4096, 256, 0, stream>>>(vv, hq0, hq1, xl, xg, x,
                                    bn_g, bn_b, bn_m, bn_v, ln1g, ln1b, out1);

  // ---- 10) FFN: gelu(out1@W1+b1) @ W2 + b2
  {
    GemmP p{}; p.A = out1; p.lda = 512; p.Bp = fw1; p.ldb = 2048; p.C = ffnh; p.ldc = 2048;
    p.M = 4096; p.N = 2048; p.K = 512; p.zdiv = 1; p.bias = fb1; p.act = 2; launch_gemm(p, 1, stream);
    GemmP q{}; q.A = ffnh; q.lda = 2048; q.Bp = fw2; q.ldb = 512; q.C = ffno; q.ldc = 512;
    q.M = 4096; q.N = 512; q.K = 2048; q.zdiv = 1; q.bias = fb2; launch_gemm(q, 1, stream);
  }

  // ---- 11) out2 = LayerNorm(out1 + ffn_out) -> d_out[0 : B*S*D)
  add_ln<<<4096, 256, 0, stream>>>(out1, ffno, ln2g, ln2b, out);
}
